// MultiHeadLatentAttention_44538810860293
// MI455X (gfx1250) — matrix operand order repaired and checked by reading, not compiled, not run on hardware
//
#include <hip/hip_runtime.h>
#include <cstdint>
#include <cstddef>

// ---------------- problem constants ----------------
#define B_DIM   2
#define S_DIM   2048
#define D_DIM   2048
#define H_DIM   16
#define NOPE_DIM 128
#define ROPE_DIM 64
#define VH_DIM  128
#define QR_DIM  512
#define KVR_DIM 512
#define QK_DIM  (NOPE_DIM + ROPE_DIM)   /* 192 */
#define MROWS   (B_DIM * S_DIM)         /* 4096 */

// ---------------- bf16 / WMMA plumbing ----------------
typedef __attribute__((ext_vector_type(16))) __bf16 v16bf;
typedef __attribute__((ext_vector_type(8)))  float  v8f;
typedef __attribute__((ext_vector_type(4)))  int    v4i;

__device__ __forceinline__ unsigned short f32_to_bf16(float f) {
  unsigned int u = __builtin_bit_cast(unsigned int, f);
  u += 0x7FFFu + ((u >> 16) & 1u);            // round-to-nearest-even
  return (unsigned short)(u >> 16);
}
__device__ __forceinline__ float bf16_to_f32(unsigned short h) {
  unsigned int u = ((unsigned int)h) << 16;
  return __builtin_bit_cast(float, u);
}
__device__ __forceinline__ v8f v8f_zero() {
  v8f z;
#pragma unroll
  for (int i = 0; i < 8; ++i) z[i] = 0.0f;
  return z;
}
__device__ __forceinline__ v8f wmma_bf16(v16bf a, v16bf b, v8f c) {
  // D = A(16x32 bf16) * B(32x16 bf16) + C(16x16 f32)
  return __builtin_amdgcn_wmma_f32_16x16x32_bf16(false, a, false, b, (short)0, c,
                                                 false, false);
}

// ---- CDNA5 async global->LDS copy (ASYNCcnt path), with safe fallback ----
#define HAVE_ASYNC_LDS (__has_builtin(__builtin_amdgcn_global_load_async_to_lds_b128))

typedef __attribute__((address_space(1))) v4i gv4i;
typedef __attribute__((address_space(3))) v4i lv4i;

__device__ __forceinline__ void cp_async16(unsigned short* lds,
                                           const unsigned short* g) {
#if HAVE_ASYNC_LDS
  __builtin_amdgcn_global_load_async_to_lds_b128((gv4i*)g, (lv4i*)lds, 0, 0);
#else
  *(uint4*)lds = *(const uint4*)g;
#endif
}
__device__ __forceinline__ void async_wait0() {
#if HAVE_ASYNC_LDS
#if __has_builtin(__builtin_amdgcn_s_wait_asynccnt)
  __builtin_amdgcn_s_wait_asynccnt(0);
#else
  asm volatile("s_wait_asynccnt 0x0" ::: "memory");
#endif
#endif
}

union FragU { v16bf v; uint4 q[2]; };

// A fragment (16M x 32K): lane L -> row = L&15 ; k-base = (L>>4)*8 ;
// two contiguous 8-elem (16B) chunks at kb and kb+16  (ISA 16-bit A layout).
__device__ __forceinline__ v16bf lds_frag_a(const unsigned short* base, int ld,
                                            int row0, int k0) {
  int lane = threadIdx.x & 31;
  const unsigned short* p =
      base + (size_t)(row0 + (lane & 15)) * ld + k0 + ((lane >> 4) << 3);
  FragU u;
  u.q[0] = *(const uint4*)(p);
  u.q[1] = *(const uint4*)(p + 16);
  return u.v;
}
// B fragment (32K x 16N) sourced from an (N x K) row-major LDS tile:
// lane L -> col = L&15 ; k-base = (L>>4)*8 ; two 8-elem chunks at kb and kb+16.
__device__ __forceinline__ v16bf lds_frag_b(const unsigned short* base, int ld,
                                            int col0, int k0) {
  int lane = threadIdx.x & 31;
  const unsigned short* p =
      base + (size_t)(col0 + (lane & 15)) * ld + k0 + ((lane >> 4) << 3);
  FragU u;
  u.q[0] = *(const uint4*)(p);
  u.q[1] = *(const uint4*)(p + 16);
  return u.v;
}

// ---------------- f32 -> bf16 convert ----------------
__global__ __launch_bounds__(256) void cvt_f32_bf16(const float* __restrict__ src,
                                                    unsigned short* __restrict__ dst,
                                                    int n) {
  for (int i = blockIdx.x * blockDim.x + threadIdx.x; i < n;
       i += blockDim.x * gridDim.x)
    dst[i] = f32_to_bf16(src[i]);
}

// ---------------- RMSNorm (f32 in, bf16 out) ----------------
__global__ __launch_bounds__(256) void rmsnorm_bf16(const float* __restrict__ x,
                                                    const float* __restrict__ w,
                                                    unsigned short* __restrict__ out,
                                                    int width) {
  __shared__ float red[8];
  int row = blockIdx.x;
  const float* xr = x + (size_t)row * width;
  float s = 0.f;
  for (int i = threadIdx.x; i < width; i += 256) { float v = xr[i]; s += v * v; }
#pragma unroll
  for (int off = 16; off >= 1; off >>= 1) s += __shfl_xor(s, off, 32);
  if ((threadIdx.x & 31) == 0) red[threadIdx.x >> 5] = s;
  __syncthreads();
  if (threadIdx.x < 8) {
    float t = red[threadIdx.x];
#pragma unroll
    for (int off = 4; off >= 1; off >>= 1) t += __shfl_xor(t, off, 32);
    if (threadIdx.x == 0) red[0] = t;
  }
  __syncthreads();
  float inv = rsqrtf(red[0] / (float)width + 1e-6f);
  for (int i = threadIdx.x; i < width; i += 256)
    out[(size_t)row * width + i] = f32_to_bf16(xr[i] * inv * w[i]);
}

// ---------------- WMMA GEMM: C[M,N] = A[M,K] * W[N,K]^T ----------------
// Block: 128 threads (4 waves). Tile: 128(M) x 64(N) x 64(K), double-buffered
// LDS filled with async global->LDS copies. Each wave: 32 rows x 64 cols ->
// 8 f32 accumulators, 16 WMMAs per K-tile.
#define GT_BM  128
#define GT_BN  64
#define GT_BK  64
#define GT_LDA 80    /* 64 + 16 pad; 160B rows keep b128 alignment */

template <bool OUT_F32>
__global__ __launch_bounds__(128) void gemm_bf16_nt(
    const unsigned short* __restrict__ A,   // M x K bf16 row-major
    const unsigned short* __restrict__ W,   // N x K bf16 row-major
    void* __restrict__ C,                   // M x N (f32 or bf16) row-major
    int M, int N, int K) {
  __shared__ __align__(16) unsigned short As[2][GT_BM * GT_LDA];
  __shared__ __align__(16) unsigned short Ws[2][GT_BN * GT_LDA];
  int tid = threadIdx.x;
  int wave = tid >> 5;
  int m0 = blockIdx.y * GT_BM;
  int n0 = blockIdx.x * GT_BN;

  v8f acc[2][4];
#pragma unroll
  for (int mf = 0; mf < 2; ++mf)
#pragma unroll
    for (int nt = 0; nt < 4; ++nt) acc[mf][nt] = v8f_zero();

  auto stage = [&](int buf, int kbase) {
    // A tile: 128x64 = 8192 elems -> 8 x 16B per thread
#pragma unroll
    for (int it = 0; it < 8; ++it) {
      int idx = (tid + it * 128) * 8;
      int r = idx >> 6, c = idx & 63;
      cp_async16(&As[buf][r * GT_LDA + c], A + (size_t)(m0 + r) * K + kbase + c);
    }
    // W tile: 64x64 = 4096 elems -> 4 x 16B per thread
#pragma unroll
    for (int it = 0; it < 4; ++it) {
      int idx = (tid + it * 128) * 8;
      int r = idx >> 6, c = idx & 63;
      cp_async16(&Ws[buf][r * GT_LDA + c], W + (size_t)(n0 + r) * K + kbase + c);
    }
  };

  int ksteps = K / GT_BK;
  stage(0, 0);
  for (int kt = 0; kt < ksteps; ++kt) {
    int buf = kt & 1;
    async_wait0();        // my async copies for buf are done
    __syncthreads();      // everyone's copies done; prior reads of buf^1 done
    if (kt + 1 < ksteps) stage(buf ^ 1, (kt + 1) * GT_BK);  // overlap with math
#pragma unroll
    for (int ks = 0; ks < 2; ++ks) {
      v16bf a0 = lds_frag_a(As[buf], GT_LDA, wave * 32, ks * 32);
      v16bf a1 = lds_frag_a(As[buf], GT_LDA, wave * 32 + 16, ks * 32);
#pragma unroll
      for (int nt = 0; nt < 4; ++nt) {
        v16bf bfr = lds_frag_b(Ws[buf], GT_LDA, nt * 16, ks * 32);
        acc[0][nt] = wmma_bf16(a0, bfr, acc[0][nt]);
        acc[1][nt] = wmma_bf16(a1, bfr, acc[1][nt]);
      }
    }
  }
  // epilogue: C frag -> row = r + (lane>>4)*8, col = lane&15
  int lane = tid & 31;
  int halfsel = (lane >> 4) << 3;
  int coll = lane & 15;
#pragma unroll
  for (int mf = 0; mf < 2; ++mf) {
#pragma unroll
    for (int nt = 0; nt < 4; ++nt) {
#pragma unroll
      for (int r = 0; r < 8; ++r) {
        size_t row = (size_t)m0 + wave * 32 + mf * 16 + halfsel + r;
        size_t col = (size_t)n0 + nt * 16 + coll;
        float v = acc[mf][nt][r];
        if (OUT_F32) ((float*)C)[row * N + col] = v;
        else         ((unsigned short*)C)[row * N + col] = f32_to_bf16(v);
      }
    }
  }
}

// ---------------- QKV assembly: RoPE + layout ----------------
// q,k: (B,H,S,192) bf16 ; vT: (B,H,VH,S) bf16 (transposed for PV B-operand)
__global__ __launch_bounds__(128) void build_qkv(
    const unsigned short* __restrict__ qn,   // (B*S, H*128)
    const unsigned short* __restrict__ qr,   // (B*S, H*64)
    const unsigned short* __restrict__ kn,   // (B*S, H*128)
    const unsigned short* __restrict__ vtmp, // (B*S, H*128)
    const unsigned short* __restrict__ kr,   // (B*S, 64)
    const float* __restrict__ freqs,         // (S, 32, 2) cos/sin
    unsigned short* __restrict__ q,
    unsigned short* __restrict__ k,
    unsigned short* __restrict__ vT) {
  int s = blockIdx.x, h = blockIdx.y, b = blockIdx.z;
  int t = threadIdx.x;
  size_t rowBS = (size_t)b * S_DIM + s;
  size_t bh = (size_t)b * H_DIM + h;
  size_t qkbase = (bh * S_DIM + s) * QK_DIM;
  q[qkbase + t] = qn[rowBS * (H_DIM * NOPE_DIM) + h * NOPE_DIM + t];
  k[qkbase + t] = kn[rowBS * (H_DIM * NOPE_DIM) + h * NOPE_DIM + t];
  vT[(bh * VH_DIM + t) * S_DIM + s] =
      vtmp[rowBS * (H_DIM * VH_DIM) + h * VH_DIM + t];
  if (t < ROPE_DIM / 2) {
    float c  = freqs[(size_t)s * ROPE_DIM + 2 * t + 0];
    float sn = freqs[(size_t)s * ROPE_DIM + 2 * t + 1];
    float x0 = bf16_to_f32(qr[rowBS * (H_DIM * ROPE_DIM) + h * ROPE_DIM + 2 * t]);
    float x1 = bf16_to_f32(qr[rowBS * (H_DIM * ROPE_DIM) + h * ROPE_DIM + 2 * t + 1]);
    q[qkbase + NOPE_DIM + 2 * t]     = f32_to_bf16(x0 * c - x1 * sn);
    q[qkbase + NOPE_DIM + 2 * t + 1] = f32_to_bf16(x0 * sn + x1 * c);
    float k0 = bf16_to_f32(kr[rowBS * ROPE_DIM + 2 * t]) * (1.0f / (float)H_DIM);
    float k1 = bf16_to_f32(kr[rowBS * ROPE_DIM + 2 * t + 1]) * (1.0f / (float)H_DIM);
    k[qkbase + NOPE_DIM + 2 * t]     = f32_to_bf16(k0 * c - k1 * sn);
    k[qkbase + NOPE_DIM + 2 * t + 1] = f32_to_bf16(k0 * sn + k1 * c);
  }
}

// ---------------- causal flash attention (WMMA) ----------------
#define FA_LDQ 200   /* 192 + 8 pad, 400B rows (16B multiple) */
#define FA_LDV 72    /* 64 + 8 pad, 144B rows */

__global__ __launch_bounds__(128) void flash_attn(
    const unsigned short* __restrict__ q,    // (B,H,S,192)
    const unsigned short* __restrict__ k,    // (B,H,S,192)
    const unsigned short* __restrict__ vT,   // (B,H,VH,S)
    unsigned short* __restrict__ attn_out) { // (B,S,H*VH) bf16
  __shared__ __align__(16) unsigned short Qs[64 * FA_LDQ];
  __shared__ __align__(16) unsigned short Ks[64 * FA_LDQ];
  __shared__ __align__(16) unsigned short Vt[128 * FA_LDV];
  __shared__ __align__(16) unsigned short Ps[4][16 * FA_LDV];
  int qt = blockIdx.x, h = blockIdx.y, b = blockIdx.z;
  int tid = threadIdx.x, wave = tid >> 5, lane = tid & 31;
  int halfsel = (lane >> 4) << 3;
  int coll = lane & 15;
  size_t bh = (size_t)b * H_DIM + h;

  // stage Q tile (64 x 192) once, asynchronously
  const unsigned short* Qg = q + (bh * S_DIM + (size_t)qt * 64) * QK_DIM;
#pragma unroll
  for (int it = 0; it < 12; ++it) {
    int idx = (tid + it * 128) * 8;
    int r = idx / QK_DIM, c = idx % QK_DIM;
    cp_async16(&Qs[r * FA_LDQ + c], Qg + (size_t)r * QK_DIM + c);
  }

  float m_i[8], l_i[8];
  v8f o[8];
#pragma unroll
  for (int r = 0; r < 8; ++r) { m_i[r] = -1e30f; l_i[r] = 0.f; }
#pragma unroll
  for (int dt = 0; dt < 8; ++dt) o[dt] = v8f_zero();

  const float scale = 0.07216878364870323f;  // 1/sqrt(192)
  int q_row_g = qt * 64 + wave * 16 + halfsel;

  for (int kt = 0; kt <= qt; ++kt) {   // causal: only tiles at/below diagonal
    const unsigned short* Kg = k + (bh * S_DIM + (size_t)kt * 64) * QK_DIM;
#pragma unroll
    for (int it = 0; it < 12; ++it) {
      int idx = (tid + it * 128) * 8;
      int r = idx / QK_DIM, c = idx % QK_DIM;
      cp_async16(&Ks[r * FA_LDQ + c], Kg + (size_t)r * QK_DIM + c);
    }
    const unsigned short* Vg = vT + bh * VH_DIM * S_DIM + (size_t)kt * 64;
#pragma unroll
    for (int it = 0; it < 8; ++it) {
      int idx = (tid + it * 128) * 8;
      int d = idx / 64, c = idx & 63;
      cp_async16(&Vt[d * FA_LDV + c], Vg + (size_t)d * S_DIM + c);
    }
    async_wait0();
    __syncthreads();

    // S = Q * K^T  (16x64 per wave, 6 k-steps over d=192)
    v8f sc[4];
#pragma unroll
    for (int i = 0; i < 4; ++i) sc[i] = v8f_zero();
#pragma unroll
    for (int ks = 0; ks < 6; ++ks) {
      v16bf af = lds_frag_a(Qs, FA_LDQ, wave * 16, ks * 32);
#pragma unroll
      for (int nt = 0; nt < 4; ++nt) {
        v16bf bf = lds_frag_b(Ks, FA_LDQ, nt * 16, ks * 32);
        sc[nt] = wmma_bf16(af, bf, sc[nt]);
      }
    }

    bool diag = (kt == qt);
    // online softmax per query row (row lives in 16 lanes of a half-wave)
#pragma unroll
    for (int r = 0; r < 8; ++r) {
      int qg = q_row_g + r;
      float vals[4];
      float rowmax = -1e30f;
#pragma unroll
      for (int nt = 0; nt < 4; ++nt) {
        float v = sc[nt][r] * scale;
        int kg = kt * 64 + nt * 16 + coll;
        if (diag && kg > qg) v = -1e30f;
        vals[nt] = v;
        rowmax = fmaxf(rowmax, v);
      }
#pragma unroll
      for (int off = 8; off >= 1; off >>= 1)
        rowmax = fmaxf(rowmax, __shfl_xor(rowmax, off, 32));
      float mn = fmaxf(m_i[r], rowmax);
      float corr = __expf(m_i[r] - mn);
      m_i[r] = mn;
      float rs = 0.f;
#pragma unroll
      for (int nt = 0; nt < 4; ++nt) {
        float p = __expf(vals[nt] - mn);
        rs += p;
        Ps[wave][(r + halfsel) * FA_LDV + nt * 16 + coll] = f32_to_bf16(p);
      }
#pragma unroll
      for (int off = 8; off >= 1; off >>= 1) rs += __shfl_xor(rs, off, 32);
      l_i[r] = l_i[r] * corr + rs;
#pragma unroll
      for (int dt = 0; dt < 8; ++dt) o[dt][r] *= corr;
    }

    // O += P * V   (P: 16x64 from per-wave LDS scratch; V^T tile gives B frags)
#pragma unroll
    for (int kk = 0; kk < 2; ++kk) {
      v16bf ap = lds_frag_a(Ps[wave], FA_LDV, 0, kk * 32);
#pragma unroll
      for (int dt = 0; dt < 8; ++dt) {
        v16bf bv = lds_frag_b(Vt, FA_LDV, dt * 16, kk * 32);
        o[dt] = wmma_bf16(ap, bv, o[dt]);
      }
    }
    __syncthreads();   // all reads of Ks/Vt done before next stage
  }

  // write (B,S,H*VH) bf16 for the output projection
#pragma unroll
  for (int dt = 0; dt < 8; ++dt) {
#pragma unroll
    for (int r = 0; r < 8; ++r) {
      size_t sg = (size_t)qt * 64 + wave * 16 + halfsel + r;
      size_t col = (size_t)h * VH_DIM + dt * 16 + coll;
      attn_out[((size_t)b * S_DIM + sg) * (H_DIM * VH_DIM) + col] =
          f32_to_bf16(o[dt][r] / l_i[r]);
    }
  }
}

// ---------------- host-side launch ----------------
extern "C" void kernel_launch(void* const* d_in, const int* in_sizes, int n_in,
                              void* d_out, int out_size, void* d_ws, size_t ws_size,
                              hipStream_t stream) {
  (void)in_sizes; (void)n_in; (void)out_size; (void)ws_size;
  const float* x      = (const float*)d_in[0];
  /* d_in[1] = mask (unused; causal handled analytically) */
  const float* freqs  = (const float*)d_in[2];
  const float* w_cq   = (const float*)d_in[3];
  const float* w_qn   = (const float*)d_in[4];
  const float* w_dqn  = (const float*)d_in[5];
  const float* w_dqr  = (const float*)d_in[6];
  const float* w_ckv  = (const float*)d_in[7];
  const float* w_kvn  = (const float*)d_in[8];
  const float* w_dkn  = (const float*)d_in[9];
  const float* w_dv   = (const float*)d_in[10];
  const float* w_kr   = (const float*)d_in[11];
  const float* w_pr   = (const float*)d_in[12];
  float* out = (float*)d_out;

  char* wp = (char*)d_ws;
  auto alloc = [&](size_t n_elems, size_t esz) -> void* {
    void* p = (void*)wp;
    wp += (n_elems * esz + 255) & ~(size_t)255;
    return p;
  };
  unsigned short* x_bf    = (unsigned short*)alloc((size_t)MROWS * D_DIM, 2);
  unsigned short* wcq_bf  = (unsigned short*)alloc((size_t)QR_DIM * D_DIM, 2);
  unsigned short* wdqn_bf = (unsigned short*)alloc((size_t)H_DIM * NOPE_DIM * QR_DIM, 2);
  unsigned short* wdqr_bf = (unsigned short*)alloc((size_t)H_DIM * ROPE_DIM * QR_DIM, 2);
  unsigned short* wckv_bf = (unsigned short*)alloc((size_t)KVR_DIM * D_DIM, 2);
  unsigned short* wdkn_bf = (unsigned short*)alloc((size_t)H_DIM * NOPE_DIM * KVR_DIM, 2);
  unsigned short* wdv_bf  = (unsigned short*)alloc((size_t)H_DIM * VH_DIM * KVR_DIM, 2);
  unsigned short* wkr_bf  = (unsigned short*)alloc((size_t)ROPE_DIM * D_DIM, 2);
  unsigned short* wpr_bf  = (unsigned short*)alloc((size_t)D_DIM * H_DIM * VH_DIM, 2);
  float*          cq_f    = (float*)alloc((size_t)MROWS * QR_DIM, 4);
  float*          ckv_f   = (float*)alloc((size_t)MROWS * KVR_DIM, 4);
  unsigned short* nq_bf   = (unsigned short*)alloc((size_t)MROWS * QR_DIM, 2);
  unsigned short* nkv_bf  = (unsigned short*)alloc((size_t)MROWS * KVR_DIM, 2);
  unsigned short* qn_tmp  = (unsigned short*)alloc((size_t)MROWS * H_DIM * NOPE_DIM, 2);
  unsigned short* qr_tmp  = (unsigned short*)alloc((size_t)MROWS * H_DIM * ROPE_DIM, 2);
  unsigned short* kn_tmp  = (unsigned short*)alloc((size_t)MROWS * H_DIM * NOPE_DIM, 2);
  unsigned short* v_tmp   = (unsigned short*)alloc((size_t)MROWS * H_DIM * VH_DIM, 2);
  unsigned short* kr_tmp  = (unsigned short*)alloc((size_t)MROWS * ROPE_DIM, 2);
  unsigned short* q_buf   = (unsigned short*)alloc((size_t)B_DIM * H_DIM * S_DIM * QK_DIM, 2);
  unsigned short* k_buf   = (unsigned short*)alloc((size_t)B_DIM * H_DIM * S_DIM * QK_DIM, 2);
  unsigned short* vT_buf  = (unsigned short*)alloc((size_t)B_DIM * H_DIM * VH_DIM * S_DIM, 2);
  unsigned short* attn_bf = (unsigned short*)alloc((size_t)MROWS * H_DIM * VH_DIM, 2);

  // 1) bf16 conversions
  cvt_f32_bf16<<<4096, 256, 0, stream>>>(x, x_bf, MROWS * D_DIM);
  cvt_f32_bf16<<<1024, 256, 0, stream>>>(w_cq,  wcq_bf,  QR_DIM * D_DIM);
  cvt_f32_bf16<<<1024, 256, 0, stream>>>(w_dqn, wdqn_bf, H_DIM * NOPE_DIM * QR_DIM);
  cvt_f32_bf16<<<512,  256, 0, stream>>>(w_dqr, wdqr_bf, H_DIM * ROPE_DIM * QR_DIM);
  cvt_f32_bf16<<<1024, 256, 0, stream>>>(w_ckv, wckv_bf, KVR_DIM * D_DIM);
  cvt_f32_bf16<<<1024, 256, 0, stream>>>(w_dkn, wdkn_bf, H_DIM * NOPE_DIM * KVR_DIM);
  cvt_f32_bf16<<<1024, 256, 0, stream>>>(w_dv,  wdv_bf,  H_DIM * VH_DIM * KVR_DIM);
  cvt_f32_bf16<<<128,  256, 0, stream>>>(w_kr,  wkr_bf,  ROPE_DIM * D_DIM);
  cvt_f32_bf16<<<4096, 256, 0, stream>>>(w_pr,  wpr_bf,  D_DIM * H_DIM * VH_DIM);

  // 2) compression projections (+ shared rope key)
  gemm_bf16_nt<true ><<<dim3(QR_DIM / GT_BN,  MROWS / GT_BM), 128, 0, stream>>>(
      x_bf, wcq_bf, cq_f, MROWS, QR_DIM, D_DIM);
  gemm_bf16_nt<true ><<<dim3(KVR_DIM / GT_BN, MROWS / GT_BM), 128, 0, stream>>>(
      x_bf, wckv_bf, ckv_f, MROWS, KVR_DIM, D_DIM);
  gemm_bf16_nt<false><<<dim3(ROPE_DIM / GT_BN, MROWS / GT_BM), 128, 0, stream>>>(
      x_bf, wkr_bf, kr_tmp, MROWS, ROPE_DIM, D_DIM);

  // 3) RMSNorm
  rmsnorm_bf16<<<MROWS, 256, 0, stream>>>(cq_f,  w_qn,  nq_bf,  QR_DIM);
  rmsnorm_bf16<<<MROWS, 256, 0, stream>>>(ckv_f, w_kvn, nkv_bf, KVR_DIM);

  // 4) decompression projections
  gemm_bf16_nt<false><<<dim3(H_DIM * NOPE_DIM / GT_BN, MROWS / GT_BM), 128, 0, stream>>>(
      nq_bf, wdqn_bf, qn_tmp, MROWS, H_DIM * NOPE_DIM, QR_DIM);
  gemm_bf16_nt<false><<<dim3(H_DIM * ROPE_DIM / GT_BN, MROWS / GT_BM), 128, 0, stream>>>(
      nq_bf, wdqr_bf, qr_tmp, MROWS, H_DIM * ROPE_DIM, QR_DIM);
  gemm_bf16_nt<false><<<dim3(H_DIM * NOPE_DIM / GT_BN, MROWS / GT_BM), 128, 0, stream>>>(
      nkv_bf, wdkn_bf, kn_tmp, MROWS, H_DIM * NOPE_DIM, KVR_DIM);
  gemm_bf16_nt<false><<<dim3(H_DIM * VH_DIM / GT_BN, MROWS / GT_BM), 128, 0, stream>>>(
      nkv_bf, wdv_bf, v_tmp, MROWS, H_DIM * VH_DIM, KVR_DIM);

  // 5) RoPE + layout for attention
  build_qkv<<<dim3(S_DIM, H_DIM, B_DIM), 128, 0, stream>>>(
      qn_tmp, qr_tmp, kn_tmp, v_tmp, kr_tmp, freqs, q_buf, k_buf, vT_buf);

  // 6) causal flash attention
  flash_attn<<<dim3(S_DIM / 64, H_DIM, B_DIM), 128, 0, stream>>>(
      q_buf, k_buf, vT_buf, attn_bf);

  // 7) output projection -> f32 d_out
  gemm_bf16_nt<true><<<dim3(D_DIM / GT_BN, MROWS / GT_BM), 128, 0, stream>>>(
      attn_bf, wpr_bf, out, MROWS, D_DIM, H_DIM * VH_DIM);
}
